// Encoder_19009525252640
// MI455X (gfx1250) — compile-verified, matrix operand order checked by reading, not run on hardware
//
#include <hip/hip_runtime.h>

#define N_NODES   10000
#define N_GRAPHS  64
#define N_LAYERS  10
#define HID       256
#define EDGE_DIM  7
#define N_EDGES   320000
#define EPS       1e-5f
#define NEG_SLOPE 0.15f

typedef float  v8f   __attribute__((ext_vector_type(8)));
typedef __bf16 v16bf __attribute__((ext_vector_type(16)));

union Frag16 { unsigned int u[8]; v16bf v; };
union FragB  { uint4 q[2]; v16bf v; };

// round-to-nearest-even float -> bf16 bits
__device__ __forceinline__ unsigned short f2bf(float f) {
  unsigned int u = __float_as_uint(f);
  u += 0x7FFFu + ((u >> 16) & 1u);
  return (unsigned short)(u >> 16);
}

// ---------------------------------------------------------------------------
// Precompute: pack nw1 (bf16) and combined C = ew2 @ nw2 (bf16) into the
// WMMA B-fragment layout: for tile (kt,nt), lane holds column n = nt*16+lane%16,
// 16 K-values with pair interleave (+8 K offset for lanes 16..31).
// packed index p = ((kt*16+nt)*32 + lane)*16 + i
// ---------------------------------------------------------------------------
__global__ void k_pack(const float* __restrict__ nw,   // [10,512,256]
                       const float* __restrict__ ew2,  // [10,256,256]
                       unsigned short* __restrict__ pkN,
                       unsigned short* __restrict__ pkC)
{
  int tid = blockIdx.x * blockDim.x + threadIdx.x;
  if (tid >= N_LAYERS * HID * HID) return;
  int l    = tid >> 16;
  int p    = tid & 65535;
  int i    = p & 15;
  int lane = (p >> 4) & 31;
  int nt   = (p >> 9) & 15;
  int kt   = p >> 13;
  int v = i >> 1, h = i & 1;
  int k = kt * 32 + (((v < 4) ? v : v + 4) + ((lane >= 16) ? 4 : 0)) * 2 + h;
  int n = nt * 16 + (lane & 15);
  const float* nw1 = nw + (long)l * 2 * HID * HID;
  const float* nw2 = nw1 + (long)HID * HID;
  pkN[tid] = f2bf(nw1[k * HID + n]);
  float s = 0.f;
  const float* e2 = ew2 + (long)l * HID * HID + (long)k * HID;
  for (int j = 0; j < HID; ++j) s += e2[j] * nw2[j * HID + n];
  pkC[tid] = f2bf(s);
}

// bias_comb[l][n] = eb2[l] @ nw2[l][:,n] + nb[l][n]
__global__ void k_bias(const float* __restrict__ nw, const float* __restrict__ eb2,
                       const float* __restrict__ nb, float* __restrict__ bias)
{
  int tid = blockIdx.x * blockDim.x + threadIdx.x;
  if (tid >= N_LAYERS * HID) return;
  int l = tid >> 8, n = tid & 255;
  const float* nw2 = nw + ((long)l * 2 + 1) * HID * HID;
  const float* b2  = eb2 + l * HID;
  float s = nb[tid];
  for (int j = 0; j < HID; ++j) s += b2[j] * nw2[j * HID + n];
  bias[tid] = s;
}

__global__ void k_init_x(const float* __restrict__ node_emb,
                         float* __restrict__ x, unsigned short* __restrict__ xbf)
{
  int i = blockIdx.x * blockDim.x + threadIdx.x;
  if (i >= N_NODES * HID) return;
  float v = node_emb[i & 255];
  x[i] = v;
  xbf[i] = f2bf(v);
}

__global__ void k_counts(const int* __restrict__ batch, float* __restrict__ counts)
{
  int n = blockIdx.x * blockDim.x + threadIdx.x;
  if (n < N_NODES) atomicAdd(&counts[batch[n]], 1.f);
}

// ---------------------------------------------------------------------------
// Fused per-layer message kernel:
//  block = 64 edges x 256 cols, 128 threads = 4 waves.
//  Each wave owns 4 N-tiles and iterates 4 M-tiles -> 4x B-fragment reuse,
//  32 independent WMMAs per K-tile to hide global/L2 latency.
//  Stages x[dst] (bf16) + computes h1 = leakyrelu(attr@ew1+eb1) (bf16) in LDS,
//  then m = x[dst]@nw1 + h1@C + bias via v_wmma_f32_16x16x32_bf16,
//  scatter-adds rows into agg with f32 global atomics.
// ---------------------------------------------------------------------------
#define EDGES_PER_BLK 64

__global__ __launch_bounds__(128)
void k_msg_gemm(const float* __restrict__ edge_attr,      // [E,7]
                const float* __restrict__ ew1,            // [7,256] (layer slice)
                const float* __restrict__ eb1,            // [256]
                const unsigned short* __restrict__ pkN,   // packed nw1 bf16 (layer)
                const unsigned short* __restrict__ pkC,   // packed C bf16 (layer)
                const float* __restrict__ bias,           // [256]
                const unsigned short* __restrict__ xbf,   // [N_NODES,256] bf16
                const int* __restrict__ dst,              // [E]
                float* __restrict__ agg)                  // [N_NODES,256]
{
  __shared__ __align__(16) unsigned short sm_x[EDGES_PER_BLK * HID];
  __shared__ __align__(16) unsigned short sm_h[EDGES_PER_BLK * HID];
  __shared__ float sm_attr[EDGES_PER_BLK][8];
  __shared__ int   sm_dst[EDGES_PER_BLK];

  const int t  = threadIdx.x;
  const int e0 = blockIdx.x * EDGES_PER_BLK;

  if (t < EDGES_PER_BLK) {
    sm_dst[t] = dst[e0 + t];
#pragma unroll
    for (int d = 0; d < EDGE_DIM; ++d)
      sm_attr[t][d] = edge_attr[(long)(e0 + t) * EDGE_DIM + d];
  }
  __syncthreads();

  {  // gather x rows (bf16) into LDS: thread t copies 256B of row t/2
    int r = t >> 1, seg = t & 1;
    const uint4* src = (const uint4*)(xbf + (long)sm_dst[r] * HID);
    uint4* dstp = (uint4*)(sm_x + r * HID);
#pragma unroll
    for (int q = 0; q < 16; ++q) dstp[seg * 16 + q] = src[seg * 16 + q];
  }
  {  // h1 tile: 7-dim edge MLP layer 1 + LeakyReLU, bf16 into LDS
    int r = t >> 1, c0 = (t & 1) * 128;
#pragma unroll 1
    for (int cc = 0; cc < 128; ++cc) {
      int c = c0 + cc;
      float a = eb1[c];
#pragma unroll
      for (int d = 0; d < EDGE_DIM; ++d) a += sm_attr[r][d] * ew1[d * HID + c];
      a = (a > 0.f) ? a : a * NEG_SLOPE;
      sm_h[r * HID + c] = f2bf(a);
    }
  }
  __syncthreads();

  const int wave = t >> 5;
  const int lane = t & 31;
  const int row  = lane & 15;
  const int hi4  = (lane >= 16) ? 4 : 0;

  v8f acc[4][4];  // [m-tile][n-tile]
#pragma unroll
  for (int mt = 0; mt < 4; ++mt)
#pragma unroll
    for (int j = 0; j < 4; ++j)
#pragma unroll
      for (int r = 0; r < 8; ++r) acc[mt][j][r] = 0.f;

#pragma unroll 1
  for (int kt = 0; kt < 8; ++kt) {
    // A fragments for all 4 M-tiles, both matrices (held in registers)
    Frag16 ax[4], ah[4];
#pragma unroll
    for (int mt = 0; mt < 4; ++mt) {
      const unsigned int* xrow =
          (const unsigned int*)(sm_x + (mt * 16 + row) * HID);
      const unsigned int* hrow =
          (const unsigned int*)(sm_h + (mt * 16 + row) * HID);
#pragma unroll
      for (int v = 0; v < 8; ++v) {
        int off = kt * 16 + ((v < 4) ? v : v + 4) + hi4;  // dword offset (K pair)
        ax[mt].u[v] = xrow[off];
        ah[mt].u[v] = hrow[off];
      }
    }
#pragma unroll
    for (int j = 0; j < 4; ++j) {
      int  nt = wave * 4 + j;
      long fb = ((long)(kt * 16 + nt) * 32 + lane) * 16;
      FragB bx, bh;
      bx.q[0] = ((const uint4*)(pkN + fb))[0];
      bx.q[1] = ((const uint4*)(pkN + fb))[1];
      bh.q[0] = ((const uint4*)(pkC + fb))[0];
      bh.q[1] = ((const uint4*)(pkC + fb))[1];
#pragma unroll
      for (int mt = 0; mt < 4; ++mt) {
        acc[mt][j] = __builtin_amdgcn_wmma_f32_16x16x32_bf16(
            false, ax[mt].v, false, bx.v, (short)0, acc[mt][j], false, false);
        acc[mt][j] = __builtin_amdgcn_wmma_f32_16x16x32_bf16(
            false, ah[mt].v, false, bh.v, (short)0, acc[mt][j], false, false);
      }
    }
  }

  // epilogue: bias + atomic scatter. C/D layout: lane<16: M=r,N=lane ; lane>=16: M=r+8
  const int n     = lane & 15;
  const int mbase = (lane >= 16) ? 8 : 0;
#pragma unroll
  for (int mt = 0; mt < 4; ++mt) {
#pragma unroll
    for (int j = 0; j < 4; ++j) {
      int col = (wave * 4 + j) * 16 + n;
      float b = bias[col];
#pragma unroll
      for (int r = 0; r < 8; ++r) {
        int m = mt * 16 + mbase + r;
        atomicAdd(&agg[(long)sm_dst[m] * HID + col], acc[mt][j][r] + b);
      }
    }
  }
}

// ---------------------------------------------------------------------------
// GraphNorm (3 passes) + pooling
// ---------------------------------------------------------------------------
__global__ void k_gn1(float* __restrict__ agg, const float* __restrict__ x,
                      const int* __restrict__ batch, float* __restrict__ meanbuf)
{
  int i = blockIdx.x * blockDim.x + threadIdx.x;
  if (i >= N_NODES * HID) return;
  int n = i >> 8, c = i & 255;
  float h = agg[i] + x[i];
  agg[i] = h;
  atomicAdd(&meanbuf[batch[n] * HID + c], h);
}

__global__ void k_gn2(float* __restrict__ agg, const float* __restrict__ meanbuf,
                      const float* __restrict__ counts, const float* __restrict__ gms,
                      const int* __restrict__ batch, float* __restrict__ varbuf)
{
  int i = blockIdx.x * blockDim.x + threadIdx.x;
  if (i >= N_NODES * HID) return;
  int n = i >> 8, c = i & 255;
  int g = batch[n];
  float mean = meanbuf[g * HID + c] / counts[g];
  float o = agg[i] - gms[c] * mean;
  agg[i] = o;
  atomicAdd(&varbuf[g * HID + c], o * o);
}

__global__ void k_gn3(const float* __restrict__ agg, const float* __restrict__ varbuf,
                      const float* __restrict__ counts, const float* __restrict__ gw,
                      const float* __restrict__ gb, const int* __restrict__ batch,
                      float* __restrict__ x, unsigned short* __restrict__ xbf)
{
  int i = blockIdx.x * blockDim.x + threadIdx.x;
  if (i >= N_NODES * HID) return;
  int n = i >> 8, c = i & 255;
  int g = batch[n];
  float var = varbuf[g * HID + c] / counts[g];
  float v = gw[c] * agg[i] * rsqrtf(var + EPS) + gb[c];
  x[i] = v;
  xbf[i] = f2bf(v);
}

__global__ void k_pool_add(const float* __restrict__ x, const int* __restrict__ batch,
                           float* __restrict__ out)
{
  int i = blockIdx.x * blockDim.x + threadIdx.x;
  if (i >= N_NODES * HID) return;
  int n = i >> 8, c = i & 255;
  atomicAdd(&out[batch[n] * HID + c], x[i]);
}

__global__ void k_pool_div(float* __restrict__ out, const float* __restrict__ counts)
{
  int i = blockIdx.x * blockDim.x + threadIdx.x;
  if (i >= N_GRAPHS * HID) return;
  out[i] /= counts[i >> 8];
}

// ---------------------------------------------------------------------------
extern "C" void kernel_launch(void* const* d_in, const int* in_sizes, int n_in,
                              void* d_out, int out_size, void* d_ws, size_t ws_size,
                              hipStream_t stream)
{
  const float* node_emb  = (const float*)d_in[0];
  const float* edge_attr = (const float*)d_in[1];
  const float* ew1       = (const float*)d_in[2];
  const float* eb1       = (const float*)d_in[3];
  const float* ew2       = (const float*)d_in[4];
  const float* eb2       = (const float*)d_in[5];
  const float* nw        = (const float*)d_in[6];
  const float* nb        = (const float*)d_in[7];
  const float* gw        = (const float*)d_in[8];
  const float* gb        = (const float*)d_in[9];
  const float* gms       = (const float*)d_in[10];
  const int*   edge_index = (const int*)d_in[11];
  const int*   batch      = (const int*)d_in[12];
  const int*   dst = edge_index + N_EDGES;  // edge_index[1]
  float* out = (float*)d_out;

  char* ws = (char*)d_ws;
  size_t off = 0;
  auto alloc = [&](size_t bytes) -> char* {
    char* p = ws + off;
    off = (off + bytes + 255) & ~(size_t)255;
    return p;
  };
  unsigned short* pkN     = (unsigned short*)alloc((size_t)N_LAYERS * HID * HID * 2);
  unsigned short* pkC     = (unsigned short*)alloc((size_t)N_LAYERS * HID * HID * 2);
  float*          biasC   = (float*)alloc((size_t)N_LAYERS * HID * 4);
  float*          x       = (float*)alloc((size_t)N_NODES * HID * 4);
  unsigned short* xbf     = (unsigned short*)alloc((size_t)N_NODES * HID * 2);
  float*          agg     = (float*)alloc((size_t)N_NODES * HID * 4);
  float*          counts  = (float*)alloc((size_t)N_GRAPHS * 4);
  float*          meanbuf = (float*)alloc((size_t)N_GRAPHS * HID * 4);
  float*          varbuf  = (float*)alloc((size_t)N_GRAPHS * HID * 4);

  // Precompute: packed weights, fused bias, counts, x0
  k_pack<<<(N_LAYERS * HID * HID + 255) / 256, 256, 0, stream>>>(nw, ew2, pkN, pkC);
  k_bias<<<(N_LAYERS * HID + 255) / 256, 256, 0, stream>>>(nw, eb2, nb, biasC);
  hipMemsetAsync(counts, 0, (size_t)N_GRAPHS * 4, stream);
  k_counts<<<(N_NODES + 255) / 256, 256, 0, stream>>>(batch, counts);
  k_init_x<<<(N_NODES * HID + 255) / 256, 256, 0, stream>>>(node_emb, x, xbf);

  for (int l = 0; l < N_LAYERS; ++l) {
    hipMemsetAsync(agg, 0, (size_t)N_NODES * HID * 4, stream);
    hipMemsetAsync(meanbuf, 0, (size_t)N_GRAPHS * HID * 4, stream);
    hipMemsetAsync(varbuf, 0, (size_t)N_GRAPHS * HID * 4, stream);

    k_msg_gemm<<<N_EDGES / EDGES_PER_BLK, 128, 0, stream>>>(
        edge_attr,
        ew1 + (size_t)l * EDGE_DIM * HID,
        eb1 + (size_t)l * HID,
        pkN + (size_t)l * HID * HID,
        pkC + (size_t)l * HID * HID,
        biasC + (size_t)l * HID,
        xbf, dst, agg);

    int nb_blk = (N_NODES * HID + 255) / 256;
    k_gn1<<<nb_blk, 256, 0, stream>>>(agg, x, batch, meanbuf);
    k_gn2<<<nb_blk, 256, 0, stream>>>(agg, meanbuf, counts, gms + (size_t)l * HID,
                                      batch, varbuf);
    k_gn3<<<nb_blk, 256, 0, stream>>>(agg, varbuf, counts, gw + (size_t)l * HID,
                                      gb + (size_t)l * HID, batch, x, xbf);
  }

  hipMemsetAsync(out, 0, (size_t)N_GRAPHS * HID * 4, stream);
  k_pool_add<<<(N_NODES * HID + 255) / 256, 256, 0, stream>>>(x, batch, out);
  k_pool_div<<<(N_GRAPHS * HID + 255) / 256, 256, 0, stream>>>(out, counts);
}
